// RecInfer_88553635709500
// MI455X (gfx1250) — matrix operand order repaired and checked by reading, not compiled, not run on hardware
//
#include <hip/hip_runtime.h>
#include <hip/hip_bf16.h>

typedef __attribute__((ext_vector_type(16))) _Float16 v16h;
typedef __attribute__((ext_vector_type(8)))  float    v8f;
typedef __attribute__((ext_vector_type(4)))  unsigned u32x4;
typedef __attribute__((ext_vector_type(8)))  int      i32x8;
typedef __attribute__((ext_vector_type(4)))  int      i32x4;

#define B_USERS   256
#define SEQ_S     50
#define PROF_P    8
#define CTX_C     4
#define N_ITEMS   1000000
#define DIM       64
#define TOPK      100
#define KSEL      150            // top-150 unmasked, filter <=50 used later
#define BINS      1024           // key >> 22 : sign + exp8 + 1 mantissa bit
#define CAP       8192           // candidate cap per user
#define TILE_R    128            // rows per TDM tile (32 KB of f32)
#define ROWPITCH  68             // 64 data dwords + 4 pad dwords (TDM pad)
#define TILES128  ((N_ITEMS + TILE_R - 1) / TILE_R)   // 7813
#define NSLICES   64
#define TPS       ((TILES128 + NSLICES - 1) / NSLICES) // 123 tiles per slice

__device__ __forceinline__ unsigned okey(float s) {
    unsigned b = __float_as_uint(s);
    return (b & 0x80000000u) ? ~b : (b | 0x80000000u);
}

// ---- TDM: issue a 2D tile load (rows x 64 f32) items[rowBase..) -> LDS buf, padded rows
__device__ __forceinline__ void tdm_issue(const float* __restrict__ items, int rowBase,
                                          int rows, float* ldsDst) {
    const unsigned long long ga = (unsigned long long)(const void*)(items + (size_t)rowBase * DIM);
    const unsigned laddr = (unsigned)(unsigned long long)(void*)ldsDst; // low 32 bits = LDS offset
    u32x4 g0;
    g0[0] = 1u;                                             // count=1, user mode
    g0[1] = laddr;                                          // lds_addr
    g0[2] = (unsigned)ga;                                   // global_addr[31:0]
    g0[3] = (unsigned)((ga >> 32) & 0x1FFFFFFull) | (2u << 30); // global_addr[56:32] | type=2
    i32x8 g1;
    g1[0] = (int)((2u << 16) | (1u << 20) | (5u << 22) | (3u << 25)); // 4B elems, pad every 64dw by 4dw
    g1[1] = (int)(64u << 16);                               // tensor_dim0 = 64
    g1[2] = (int)((N_ITEMS & 0xFFFFu) << 16);               // tensor_dim1 lo
    g1[3] = (int)((N_ITEMS >> 16) | (64u << 16));           // tensor_dim1 hi | tile_dim0 = 64
    g1[4] = (int)(unsigned)rows;                            // tile_dim1 = rows (tile_dim2 = 0)
    g1[5] = 64;                                             // tensor_dim0_stride = 64
    g1[6] = 0;
    g1[7] = 0;
    i32x4 gz4 = {0, 0, 0, 0};
    i32x8 gz8 = {0, 0, 0, 0, 0, 0, 0, 0};
    __builtin_amdgcn_tensor_load_to_lds(g0, g1, gz4, gz4, gz8, 0);
}

// ---------------------------------------------------------------- zero ws
__global__ void k_zero(unsigned* p, int n) {
    for (int i = blockIdx.x * blockDim.x + threadIdx.x; i < n; i += gridDim.x * blockDim.x)
        p[i] = 0u;
}

// ---------------------------------------------------------------- user tower
__global__ void k_user(const int* __restrict__ profile, const int* __restrict__ context,
                       const int* __restrict__ item_indices,
                       const float* __restrict__ item_vectors,
                       const float* __restrict__ profile_emb,
                       const float* __restrict__ context_emb,
                       const float* __restrict__ W_u,
                       float* __restrict__ userv) {
    const int b = blockIdx.x;
    const int d = threadIdx.x;
    float p = 0.f, c = 0.f, it = 0.f;
    #pragma unroll
    for (int j = 0; j < PROF_P; ++j) p += profile_emb[(size_t)profile[b * PROF_P + j] * DIM + d];
    #pragma unroll
    for (int j = 0; j < CTX_C; ++j)  c += context_emb[(size_t)context[b * CTX_C + j] * DIM + d];
    for (int j = 0; j < SEQ_S; ++j)  it += item_vectors[(size_t)item_indices[b * SEQ_S + j] * DIM + d];
    __shared__ float pooled[DIM];
    pooled[d] = p * (1.0f / PROF_P) + c * (1.0f / CTX_C) + it * (1.0f / SEQ_S);
    __syncthreads();
    float u = 0.f;
    #pragma unroll
    for (int k = 0; k < DIM; ++k) u = fmaf(pooled[k], W_u[k * DIM + d], u);
    userv[b * DIM + d] = u;
}

// ---------------------------------------------------------------- A fragment build (16-bit 16x32 layout)
__device__ __forceinline__ void build_afrag(const float* __restrict__ userv,
                                            int userGroup, int lane,
                                            v16h& a0, v16h& a1) {
    const int g = lane >> 4;
    const int m = lane & 15;
    const float* uv = userv + (size_t)(userGroup * 16 + m) * DIM;
    #pragma unroll
    for (int h = 0; h < 16; ++h) {
        const int v = h >> 1, p = h & 1;
        const int K = g * 16 + 2 * v + p;
        a0[h] = (_Float16)uv[K];
        a1[h] = (_Float16)uv[32 + K];
    }
}

// ---------------------------------------------------------------- B fragment from (padded) LDS row
__device__ __forceinline__ void load_bfrag_lds(const float* __restrict__ row, int kb,
                                               v16h& b0, v16h& b1) {
    float f0[16], f1[16];
    const float4* p0 = (const float4*)(row + kb);
    const float4* p1 = (const float4*)(row + 32 + kb);
    #pragma unroll
    for (int q = 0; q < 4; ++q) {
        ((float4*)f0)[q] = p0[q];
        ((float4*)f1)[q] = p1[q];
    }
    #pragma unroll
    for (int h = 0; h < 16; ++h) { b0[h] = (_Float16)f0[h]; b1[h] = (_Float16)f1[h]; }
}

// ---------------------------------------------------------------- pass 1: TDM stream + WMMA + histogram
// grid (16 user groups [x, fast-varying -> L2 sharing], NSLICES [y]) ; block 128 (4 waves)
__global__ void k_hist(const float* __restrict__ items, const float* __restrict__ userv,
                       unsigned* __restrict__ hist) {
    __shared__ unsigned lhist[16 * BINS];                       // 64 KB
    __shared__ __align__(16) float tdbuf[2][TILE_R * ROWPITCH]; // 2 x 34 KB
    for (int i = threadIdx.x; i < 16 * BINS; i += blockDim.x) lhist[i] = 0u;

    const int lane = threadIdx.x & 31;
    const int wave = threadIdx.x >> 5;
    const int grp  = blockIdx.x;
    v16h a0, a1;
    build_afrag(userv, grp, lane, a0, a1);

    const int nlocal = lane & 15;
    const int kb = (lane >> 4) * 16;
    const int mbase = (lane >> 4) * 8;

    const int tBeg = blockIdx.y * TPS;
    const int tEnd = min(TILES128, tBeg + TPS);

    if (tBeg < tEnd) {
        if (wave == 0) {
            tdm_issue(items, tBeg * TILE_R, min(TILE_R, N_ITEMS - tBeg * TILE_R), &tdbuf[0][0]);
            __builtin_amdgcn_s_wait_tensorcnt(0);
        }
    }
    __syncthreads();

    for (int t = tBeg; t < tEnd; ++t) {
        const int cur = (t - tBeg) & 1;
        if (wave == 0 && t + 1 < tEnd)
            tdm_issue(items, (t + 1) * TILE_R, min(TILE_R, N_ITEMS - (t + 1) * TILE_R),
                      &tdbuf[cur ^ 1][0]);

        const int rows = min(TILE_R, N_ITEMS - t * TILE_R);
        const int nsub = rows >> 4;
        for (int s = wave; s < nsub; s += 4) {
            const float* lrow = &tdbuf[cur][(s * 16 + nlocal) * ROWPITCH];
            v16h b0, b1;
            load_bfrag_lds(lrow, kb, b0, b1);
            v8f c = {};
            c = __builtin_amdgcn_wmma_f32_16x16x32_f16(false, a0, false, b0, (short)0, c, false, false);
            c = __builtin_amdgcn_wmma_f32_16x16x32_f16(false, a1, false, b1, (short)0, c, false, false);
            #pragma unroll
            for (int v = 0; v < 8; ++v) {
                const unsigned u = okey(c[v]);
                atomicAdd(&lhist[(mbase + v) * BINS + (u >> 22)], 1u);
            }
        }
        if (wave == 0) __builtin_amdgcn_s_wait_tensorcnt(0);
        __syncthreads();
    }

    unsigned* gh = hist + (size_t)grp * 16 * BINS;
    for (int i = threadIdx.x; i < 16 * BINS; i += blockDim.x) {
        const unsigned v = lhist[i];
        if (v) atomicAdd(&gh[i], v);
    }
}

// ---------------------------------------------------------------- pass 2: threshold scan
__global__ void k_thresh(const unsigned* __restrict__ hist, unsigned* __restrict__ thr) {
    const int b = threadIdx.x;
    const unsigned* h = hist + (size_t)b * BINS;
    unsigned cum = 0, t = 0u;
    for (int bin = BINS - 1; bin >= 0; --bin) {
        cum += h[bin];
        if (cum >= KSEL) { t = ((unsigned)bin) << 22; break; }
    }
    thr[b] = t;
}

// ---------------------------------------------------------------- pass 3: TDM stream + WMMA + collect
__global__ void k_collect(const float* __restrict__ items, const float* __restrict__ userv,
                          const unsigned* __restrict__ thr,
                          unsigned* __restrict__ cnt,
                          unsigned long long* __restrict__ cand) {
    __shared__ __align__(16) float tdbuf[2][TILE_R * ROWPITCH];
    const int lane = threadIdx.x & 31;
    const int wave = threadIdx.x >> 5;
    const int grp  = blockIdx.x;
    v16h a0, a1;
    build_afrag(userv, grp, lane, a0, a1);

    const int mbase = (lane >> 4) * 8;
    unsigned uthr[8];
    #pragma unroll
    for (int v = 0; v < 8; ++v) uthr[v] = thr[grp * 16 + mbase + v];

    const int nlocal = lane & 15;
    const int kb = (lane >> 4) * 16;

    const int tBeg = blockIdx.y * TPS;
    const int tEnd = min(TILES128, tBeg + TPS);

    if (tBeg < tEnd) {
        if (wave == 0) {
            tdm_issue(items, tBeg * TILE_R, min(TILE_R, N_ITEMS - tBeg * TILE_R), &tdbuf[0][0]);
            __builtin_amdgcn_s_wait_tensorcnt(0);
        }
    }
    __syncthreads();

    for (int t = tBeg; t < tEnd; ++t) {
        const int cur = (t - tBeg) & 1;
        if (wave == 0 && t + 1 < tEnd)
            tdm_issue(items, (t + 1) * TILE_R, min(TILE_R, N_ITEMS - (t + 1) * TILE_R),
                      &tdbuf[cur ^ 1][0]);

        const int rows = min(TILE_R, N_ITEMS - t * TILE_R);
        const int nsub = rows >> 4;
        for (int s = wave; s < nsub; s += 4) {
            const int item = t * TILE_R + s * 16 + nlocal;
            const float* lrow = &tdbuf[cur][(s * 16 + nlocal) * ROWPITCH];
            v16h b0, b1;
            load_bfrag_lds(lrow, kb, b0, b1);
            v8f c = {};
            c = __builtin_amdgcn_wmma_f32_16x16x32_f16(false, a0, false, b0, (short)0, c, false, false);
            c = __builtin_amdgcn_wmma_f32_16x16x32_f16(false, a1, false, b1, (short)0, c, false, false);
            #pragma unroll
            for (int v = 0; v < 8; ++v) {
                const unsigned u = okey(c[v]);
                if (u >= uthr[v]) {
                    const int user = grp * 16 + mbase + v;
                    const unsigned pos = atomicAdd(&cnt[user], 1u);
                    if (pos < CAP)
                        cand[(size_t)user * CAP + pos] =
                            ((unsigned long long)u << 32) | (unsigned long long)(~(unsigned)item);
                }
            }
        }
        if (wave == 0) __builtin_amdgcn_s_wait_tensorcnt(0);
        __syncthreads();
    }
}

// ---------------------------------------------------------------- pass 4: per-user sort + mask + emit
__global__ void k_final(const unsigned long long* __restrict__ cand,
                        const unsigned* __restrict__ cnt,
                        const int* __restrict__ item_indices,
                        float* __restrict__ out) {
    __shared__ unsigned long long arr[CAP];          // 64 KB
    const int b = blockIdx.x;
    const int tid = threadIdx.x;
    const unsigned cb = min(cnt[b], (unsigned)CAP);
    for (int i = tid; i < CAP; i += blockDim.x)
        arr[i] = (i < (int)cb) ? cand[(size_t)b * CAP + i] : 0ull;
    __syncthreads();

    for (int k = 2; k <= CAP; k <<= 1) {
        for (int j = k >> 1; j > 0; j >>= 1) {
            for (int i = tid; i < CAP; i += blockDim.x) {
                const int ij = i ^ j;
                if (ij > i) {
                    const unsigned long long x = arr[i], y = arr[ij];
                    const bool up = ((i & k) == 0);
                    if (up ? (x < y) : (x > y)) { arr[i] = y; arr[ij] = x; }
                }
            }
            __syncthreads();
        }
    }

    if (tid == 0) {
        const int* used = item_indices + b * SEQ_S;
        int written = 0;
        for (int i = 0; i < CAP && written < TOPK; ++i) {
            const unsigned long long e = arr[i];
            if (e == 0ull) break;
            const unsigned idx = ~(unsigned)(e & 0xffffffffu);
            if (idx >= N_ITEMS) continue;
            bool isUsed = false;
            #pragma unroll 1
            for (int j = 0; j < SEQ_S; ++j) if ((unsigned)used[j] == idx) { isUsed = true; break; }
            if (!isUsed) out[b * TOPK + written++] = (float)idx;
        }
        while (written < TOPK) out[b * TOPK + written++] = 0.f;
    }
}

// ---------------------------------------------------------------- launcher
extern "C" void kernel_launch(void* const* d_in, const int* in_sizes, int n_in,
                              void* d_out, int out_size, void* d_ws, size_t ws_size,
                              hipStream_t stream) {
    const int*   profile      = (const int*)d_in[0];
    const int*   context      = (const int*)d_in[1];
    const int*   item_indices = (const int*)d_in[2];
    const float* item_vectors = (const float*)d_in[3];
    const float* profile_emb  = (const float*)d_in[4];
    const float* context_emb  = (const float*)d_in[5];
    const float* W_u          = (const float*)d_in[6];
    float* out = (float*)d_out;

    char* ws = (char*)d_ws;
    float*              userv = (float*)ws;                               // 64 KB
    unsigned*           hist  = (unsigned*)(ws + 65536);                  // 1 MB
    unsigned*           thr   = (unsigned*)(ws + 65536 + 1048576);        // 1 KB
    unsigned*           cnt   = (unsigned*)(ws + 65536 + 1048576 + 1024); // 1 KB
    unsigned long long* cand  = (unsigned long long*)(ws + 65536 + 1048576 + 2048); // 16 MB

    k_zero<<<dim3(512), dim3(256), 0, stream>>>(hist, B_USERS * BINS);
    k_zero<<<dim3(2), dim3(256), 0, stream>>>(cnt, B_USERS);

    k_user<<<dim3(B_USERS), dim3(DIM), 0, stream>>>(profile, context, item_indices,
                                                    item_vectors, profile_emb, context_emb,
                                                    W_u, userv);

    k_hist<<<dim3(16, NSLICES), dim3(128), 0, stream>>>(item_vectors, userv, hist);

    k_thresh<<<dim3(1), dim3(B_USERS), 0, stream>>>(hist, thr);

    k_collect<<<dim3(16, NSLICES), dim3(128), 0, stream>>>(item_vectors, userv, thr, cnt, cand);

    k_final<<<dim3(B_USERS), dim3(256), 0, stream>>>(cand, cnt, item_indices, out);
}
